// DLSTMCell_65317862637836
// MI455X (gfx1250) — matrix operand order repaired and checked by reading, not compiled, not run on hardware
//
#include <hip/hip_runtime.h>
#include <hip/hip_bf16.h>

typedef __bf16 bf16;
typedef __attribute__((ext_vector_type(16))) __bf16 v16bf;
typedef __attribute__((ext_vector_type(8)))  __bf16 v8bf;
typedef __attribute__((ext_vector_type(8)))  float  v8f;
typedef __attribute__((ext_vector_type(4)))  int    v4i;

#define N_STOCKS 2048
#define INPUT    64
#define EMB      128
#define BOT      64
#define HID      128
#define NGATES   512        // 4*HID
#define DDIM     192        // INPUT+HID
#define WDIM     98304      // DDIM*NGATES
#define K_MAIN   12288      // DDIM*BOT
#define K_TOT    12480      // K_MAIN + DDIM (b3 folded into GEMM K)

#define MB    64            // block tile M
#define NBLK  128           // block tile N (2 waves x 64)
#define KB    32            // K step (one WMMA K)
#define KSPLIT 2
#define KCH   (K_TOT / KSPLIT)   // 6240, multiple of KB

// CDNA5 async global->LDS DMA (ASYNCcnt) if the toolchain declares it.
#if __has_builtin(__builtin_amdgcn_global_load_async_to_lds_b128)
#define USE_ASYNC_LDS 1
#endif

__device__ __forceinline__ float sigf(float v) { return 1.0f / (1.0f + __expf(-v)); }

// ---------------------------------------------------------------------------
// Kernel 1: hypernetwork front-end: h2 = tanh(tanh(emb[idx]@w1^T+b1)@w2^T+b2)
// ---------------------------------------------------------------------------
__global__ __launch_bounds__(128) void k_h2(const float* __restrict__ emb,
                                            const int*   __restrict__ index,
                                            const float* __restrict__ w1,
                                            const float* __restrict__ b1,
                                            const float* __restrict__ w2,
                                            const float* __restrict__ b2,
                                            bf16* __restrict__ h2out) {
    __shared__ float sm[EMB];
    __shared__ float sh1[EMB];
    const int n = blockIdx.x;
    const int t = threadIdx.x;
    const int row = index[n];
    sm[t] = emb[(size_t)row * EMB + t];
    __syncthreads();
    float a1 = b1[t];
    const float* wr = w1 + (size_t)t * EMB;
    #pragma unroll 8
    for (int k = 0; k < EMB; ++k) a1 += sm[k] * wr[k];
    sh1[t] = tanhf(a1);
    __syncthreads();
    if (t < BOT) {
        float a2 = b2[t];
        const float* w2r = w2 + (size_t)t * EMB;
        #pragma unroll 8
        for (int k = 0; k < EMB; ++k) a2 += sh1[k] * w2r[k];
        h2out[(size_t)n * BOT + t] = (bf16)tanhf(a2);
    }
}

// ---------------------------------------------------------------------------
// Kernel 2: repack w3 (WDIM x BOT f32) -> BT (NGATES x K_TOT bf16), N-major
// ---------------------------------------------------------------------------
__global__ void k_repack(const float* __restrict__ w3,
                         const float* __restrict__ b3,
                         bf16* __restrict__ bt) {
    const long i = (long)blockIdx.x * blockDim.x + threadIdx.x;
    const long MAIN = (long)WDIM * BOT;            // 6,291,456
    if (i < MAIN) {
        int b = (int)(i & 63);
        int n = (int)((i >> 6) & 511);
        int d = (int)(i >> 15);
        bt[(long)n * K_TOT + (long)d * 64 + b] = (bf16)w3[i];
    } else {
        long j = i - MAIN;                         // [0, 512*192)
        if (j < (long)NGATES * DDIM) {
            int n = (int)(j / DDIM);
            int d = (int)(j % DDIM);
            bt[(long)n * K_TOT + K_MAIN + d] = (bf16)b3[(long)d * NGATES + n];
        }
    }
}

// ---------------------------------------------------------------------------
// Kernel 3 helpers
// ---------------------------------------------------------------------------
struct BReg { uint4 r[8]; };                   // 2 rows x 64B of the B tile
struct AReg { v8bf v[4]; };                    // one 32-wide row of the P tile

#ifdef USE_ASYNC_LDS
// param 0: int4 in addrspace(1) (global), param 1: int4 in addrspace(3) (LDS)
typedef __attribute__((address_space(1))) v4i gv4i;
typedef __attribute__((address_space(3))) v4i lv4i;

__device__ __forceinline__ void async_cp16(const bf16* g, bf16* l) {
    __builtin_amdgcn_global_load_async_to_lds_b128((gv4i*)g, (lv4i*)l, 0, 0);
}
__device__ __forceinline__ void wait_async0() {
#if __has_builtin(__builtin_amdgcn_s_wait_asynccnt)
    __builtin_amdgcn_s_wait_asynccnt(0);
#else
    asm volatile("s_wait_asynccnt 0" ::: "memory");
#endif
}
// stage 2 rows (64B each) of the B tile via DMA, no data VGPRs
__device__ __forceinline__ void stage_B_async(const bf16* __restrict__ bt_row0, int k0,
                                              bf16* __restrict__ sb, int t) {
    const bf16* g0 = bt_row0 + k0;
    const bf16* g1 = bt_row0 + K_TOT + k0;
    bf16* l0 = sb + (2 * t) * KB;
    bf16* l1 = sb + (2 * t + 1) * KB;
    #pragma unroll
    for (int j = 0; j < 4; ++j) {
        async_cp16(g0 + j * 8, l0 + j * 8);
        async_cp16(g1 + j * 8, l1 + j * 8);
    }
}
#endif

__device__ __forceinline__ BReg load_B(const bf16* __restrict__ bt_row0, int k0) {
    BReg o;
    const uint4* a = (const uint4*)(bt_row0 + k0);
    const uint4* b = (const uint4*)(bt_row0 + K_TOT + k0);
    #pragma unroll
    for (int j = 0; j < 4; ++j) { o.r[j] = a[j]; o.r[4 + j] = b[j]; }
    return o;
}
__device__ __forceinline__ void store_B(const BReg& o, bf16* __restrict__ sb, int t) {
    uint4* d0 = (uint4*)(sb + (2 * t) * KB);
    uint4* d1 = (uint4*)(sb + (2 * t + 1) * KB);
    #pragma unroll
    for (int j = 0; j < 4; ++j) { d0[j] = o.r[j]; d1[j] = o.r[4 + j]; }
}
// thread t builds row t (32 K-elements) of the P tile at tile base k0
__device__ __forceinline__ AReg build_A(const bf16* __restrict__ s_data,
                                        const bf16* __restrict__ s_h2,
                                        int r, int k0) {
    AReg o;
    if (k0 < K_MAIN) {                          // tile-uniform branch (scalar)
        const int d  = k0 >> 6;
        const int b0 = k0 & 63;
        const bf16 dv = s_data[r * DDIM + d];
        v8bf ds;
        #pragma unroll
        for (int i = 0; i < 8; ++i) ds[i] = dv;
        const v8bf* h = (const v8bf*)(s_h2 + r * BOT + b0);
        #pragma unroll
        for (int j = 0; j < 4; ++j) o.v[j] = h[j] * ds;   // v_pk_mul_bf16 path
    } else {                                    // b3 extension: P = data
        const v8bf* sd = (const v8bf*)(s_data + r * DDIM + (k0 - K_MAIN));
        #pragma unroll
        for (int j = 0; j < 4; ++j) o.v[j] = sd[j];
    }
    return o;
}
__device__ __forceinline__ void store_A(const AReg& o, bf16* __restrict__ sa, int t) {
    v8bf* d = (v8bf*)(sa + t * KB);
    #pragma unroll
    for (int j = 0; j < 4; ++j) d[j] = o.v[j];
}
__device__ __forceinline__ v16bf frag_A(const bf16* __restrict__ sa, int row, int lh) {
    const bf16* base = sa + row * KB + lh * 8;
    v8bf lo = *(const v8bf*)(base);
    v8bf hi = *(const v8bf*)(base + 16);
    return __builtin_shufflevector(lo, hi, 0, 1, 2, 3, 4, 5, 6, 7,
                                   8, 9, 10, 11, 12, 13, 14, 15);
}
__device__ __forceinline__ v16bf frag_B(const bf16* __restrict__ sb, int col, int lh) {
    const bf16* base = sb + col * KB + lh * 8;
    v8bf lo = *(const v8bf*)(base);
    v8bf hi = *(const v8bf*)(base + 16);
    return __builtin_shufflevector(lo, hi, 0, 1, 2, 3, 4, 5, 6, 7,
                                   8, 9, 10, 11, 12, 13, 14, 15);
}

// ---------------------------------------------------------------------------
// Kernel 3: gates_part[z] = P @ BT^T over K-chunk z; P built on the fly:
//   P[n, d*64+b] = data[n,d]*h2[n,b] ; P[n, 12288+d] = data[n,d]
// Block = 64 threads (2 waves), block tile 64x128, per-wave 64x64 (4x4 frags),
// double-buffered LDS (async DMA staging when available), one barrier / K step.
// ---------------------------------------------------------------------------
__global__ __launch_bounds__(64) void k_gates_gemm(const float* __restrict__ x,
                                                   const float* __restrict__ hx,
                                                   const bf16*  __restrict__ h2,
                                                   const bf16*  __restrict__ bt,
                                                   float* __restrict__ gpart) {
    __shared__ __align__(16) bf16 s_data[MB * DDIM];      // 24 KB
    __shared__ __align__(16) bf16 s_h2[MB * BOT];         //  8 KB
    __shared__ __align__(16) bf16 s_a[2][MB * KB];        //  8 KB
    __shared__ __align__(16) bf16 s_b[2][NBLK * KB];      // 16 KB

    const int t    = threadIdx.x;            // 0..63
    const int lane = t & 31;
    const int wn   = t >> 5;                 // wave -> N half (0..1)
    const int lr   = lane & 15;
    const int lh   = lane >> 4;
    const int m_blk = blockIdx.x * MB;
    const int n_blk = blockIdx.y * NBLK;
    const int kbeg  = blockIdx.z * KCH;
    float* gout = gpart + (size_t)blockIdx.z * N_STOCKS * NGATES;

    // ---- stage per-block activations (read-only thereafter) ----
    for (int i = t; i < MB * DDIM; i += 64) {
        int r = i / DDIM, d = i % DDIM;
        float v = (d < INPUT) ? x[(size_t)(m_blk + r) * INPUT + d]
                              : hx[(size_t)(m_blk + r) * HID + (d - INPUT)];
        s_data[i] = (bf16)v;
    }
    for (int i = t; i < MB * BOT; i += 64)
        s_h2[i] = h2[(size_t)m_blk * BOT + i];
    __syncthreads();

    const bf16* bt_row0 = bt + (size_t)(n_blk + 2 * t) * K_TOT;

    // ---- prologue: stage first tile into buffer 0 ----
    {
#ifdef USE_ASYNC_LDS
        stage_B_async(bt_row0, kbeg, s_b[0], t);
#else
        BReg b0 = load_B(bt_row0, kbeg);
        store_B(b0, s_b[0], t);
#endif
        AReg a0 = build_A(s_data, s_h2, t, kbeg);
        store_A(a0, s_a[0], t);
#ifdef USE_ASYNC_LDS
        wait_async0();
#endif
    }
    __syncthreads();

    v8f acc[4][4];
    #pragma unroll
    for (int mi = 0; mi < 4; ++mi)
        #pragma unroll
        for (int ni = 0; ni < 4; ++ni)
            acc[mi][ni] = (v8f){0.f, 0.f, 0.f, 0.f, 0.f, 0.f, 0.f, 0.f};

    const int niter = KCH / KB;              // 195
    int p = 0;
    for (int it = 0; it < niter; ++it) {
        const int k0 = kbeg + it * KB;
        const int k1 = k0 + KB;
        const bf16* sa = s_a[p];
        const bf16* sb = s_b[p];
        bf16* san = s_a[p ^ 1];
        bf16* sbn = s_b[p ^ 1];
        const bool nxt = (it + 1 < niter);

        // issue next B tile first so DMA/loads overlap the matrix math
#ifdef USE_ASYNC_LDS
        if (nxt) stage_B_async(bt_row0, k1, sbn, t);
#else
        BReg nb;
        if (nxt) {
            nb = load_B(bt_row0, k1);
            if (it + 2 < niter)
                __builtin_prefetch((const void*)(bt_row0 + k0 + 2 * KB), 0, 3);
        }
#endif

        // A fragments stay resident; B fragments loaded just-in-time
        v16bf af[4];
        #pragma unroll
        for (int mi = 0; mi < 4; ++mi) af[mi] = frag_A(sa, mi * 16 + lr, lh);

        #pragma unroll
        for (int ni = 0; ni < 4; ++ni) {
            v16bf bfr = frag_B(sb, wn * 64 + ni * 16 + lr, lh);
            #pragma unroll
            for (int mi = 0; mi < 4; ++mi)
                acc[mi][ni] = __builtin_amdgcn_wmma_f32_16x16x32_bf16(
                    false, af[mi], false, bfr, (short)0, acc[mi][ni], false, false);
        }

        if (nxt) {
#ifndef USE_ASYNC_LDS
            store_B(nb, sbn, t);
#endif
            AReg na = build_A(s_data, s_h2, t, k1);   // LDS-only, after the math
            store_A(na, san, t);
#ifdef USE_ASYNC_LDS
            wait_async0();
#endif
        }
        __syncthreads();
        p ^= 1;
    }

    // ---- store D per ISA C/D layout: lane = n + 16*(m>=8), vgpr = m&7 ----
    #pragma unroll
    for (int mi = 0; mi < 4; ++mi)
        #pragma unroll
        for (int ni = 0; ni < 4; ++ni) {
            int n  = n_blk + wn * 64 + ni * 16 + lr;
            int m0 = m_blk + mi * 16 + lh * 8;
            #pragma unroll
            for (int i = 0; i < 8; ++i)
                gout[(size_t)(m0 + i) * NGATES + n] = acc[mi][ni][i];
        }
}

// ---------------------------------------------------------------------------
// Kernel 4: sum split-K partials + LSTM epilogue
// ---------------------------------------------------------------------------
__global__ void k_lstm(const float* __restrict__ gpart,
                       const float* __restrict__ lstm_bias,
                       const float* __restrict__ cx,
                       float* __restrict__ out) {
    const int i = blockIdx.x * blockDim.x + threadIdx.x;   // [0, 2048*128)
    if (i >= N_STOCKS * HID) return;
    const int n = i >> 7;
    const int j = i & 127;
    const float* g0 = gpart + (size_t)n * NGATES;
    const float* g1 = g0 + (size_t)N_STOCKS * NGATES;
    float p0 = g0[j]           + g1[j];
    float p1 = g0[HID + j]     + g1[HID + j];
    float p2 = g0[2 * HID + j] + g1[2 * HID + j];
    float p3 = g0[3 * HID + j] + g1[3 * HID + j];
    float s0 = sigf(p0 + lstm_bias[j]);
    float s1 = sigf(p1 + lstm_bias[HID + j]);
    float s2 = sigf(p2 + lstm_bias[2 * HID + j]);
    float s3 = sigf(p3 + lstm_bias[3 * HID + j]);
    float it = sigf(s0), ft = sigf(s1), gt = tanhf(s2), ot = sigf(s3);
    float cy = cx[i] * ft + it * gt;
    float hy = ot * tanhf(cy);
    out[i] = hy;                      // hy block
    out[N_STOCKS * HID + i] = cy;     // cy block
}

// ---------------------------------------------------------------------------
extern "C" void kernel_launch(void* const* d_in, const int* in_sizes, int n_in,
                              void* d_out, int out_size, void* d_ws, size_t ws_size,
                              hipStream_t stream) {
    const float* x         = (const float*)d_in[0];
    const int*   index     = (const int*)  d_in[1];
    const float* hx        = (const float*)d_in[2];
    const float* cx        = (const float*)d_in[3];
    const float* emb_table = (const float*)d_in[4];
    const float* w1        = (const float*)d_in[5];
    const float* b1        = (const float*)d_in[6];
    const float* w2        = (const float*)d_in[7];
    const float* b2        = (const float*)d_in[8];
    const float* w3        = (const float*)d_in[9];
    const float* b3        = (const float*)d_in[10];
    const float* lstm_bias = (const float*)d_in[11];
    float* out = (float*)d_out;

    // workspace layout
    char* ws = (char*)d_ws;
    bf16*  h2ws = (bf16*)ws;                                   // 2048*64*2       = 256 KB
    bf16*  btws = (bf16*)(ws + (size_t)N_STOCKS * BOT * 2);    // 512*12480*2     = 12.2 MB
    float* gws  = (float*)(ws + (size_t)N_STOCKS * BOT * 2
                              + (size_t)NGATES * K_TOT * 2);   // 2*2048*512*4    = 8 MB

    // 1) hypernetwork front-end -> h2 (bf16)
    k_h2<<<N_STOCKS, 128, 0, stream>>>(emb_table, index, w1, b1, w2, b2, h2ws);

    // 2) repack w3 (+b3 extension) -> BT bf16, N-major
    {
        long total = (long)WDIM * BOT + (long)NGATES * DDIM;   // 6,389,760
        int blocks = (int)((total + 255) / 256);
        k_repack<<<blocks, 256, 0, stream>>>(w3, b3, btws);
    }

    // 3) fused hyper-weight GEMM: split-K partials (M=2048, N=512, K=12480)
    k_gates_gemm<<<dim3(N_STOCKS / MB, NGATES / NBLK, KSPLIT), 64, 0, stream>>>(
        x, hx, h2ws, btws, gws);

    // 4) partial sum + LSTM elementwise epilogue -> hy || cy
    k_lstm<<<(N_STOCKS * HID + 255) / 256, 256, 0, stream>>>(gws, lstm_bias, cx, out);
}
